// NemotronFlashDecoderLayer_89850715833068
// MI455X (gfx1250) — matrix operand order repaired and checked by reading, not compiled, not run on hardware
//
#include <hip/hip_runtime.h>
#include <stdint.h>

#define T_TOK 2048
#define HIDN  2048
#define NH    16
#define DKK   128
#define DVV   128

typedef __bf16 bf16_t;
typedef bf16_t bf16x16 __attribute__((ext_vector_type(16)));
typedef float  f32x8   __attribute__((ext_vector_type(8)));
typedef unsigned int u32x4 __attribute__((ext_vector_type(4)));
typedef unsigned int u32x8 __attribute__((ext_vector_type(8)));

#define LDS_AS __attribute__((address_space(3)))

__device__ __forceinline__ unsigned short f2bf(float x) {
  unsigned u = __float_as_uint(x);
  unsigned r = (u + 0x7FFFu + ((u >> 16) & 1u)) >> 16;  // RNE
  return (unsigned short)r;
}
__device__ __forceinline__ f32x8 zero8() {
  f32x8 z;
#pragma unroll
  for (int i = 0; i < 8; ++i) z[i] = 0.f;
  return z;
}
__device__ __forceinline__ f32x8 wmma_bf16(bf16x16 a, bf16x16 b, f32x8 c) {
  return __builtin_amdgcn_wmma_f32_16x16x32_bf16(false, a, false, b, (short)0, c,
                                                 false, false);
}

// LDS byte offset of a __shared__ object (addrspace(3) pointers are LDS offsets)
__device__ __forceinline__ unsigned lds_off(const void* p) {
  return (unsigned)(unsigned long long)(LDS_AS const void*)p;
}
// 16B async copy global -> LDS (tracked by ASYNCcnt)
__device__ __forceinline__ void async_b128(unsigned lds, const void* g) {
  unsigned long long ga = (unsigned long long)g;
  asm volatile("global_load_async_to_lds_b128 %0, %1, off" ::"v"(lds), "v"(ga)
               : "memory");
}
__device__ __forceinline__ void wait_async() {
  asm volatile("s_wait_asynccnt 0x0" ::: "memory");
}

// ---- Tensor Data Mover: 2D tile (bf16) global -> LDS, one issue per wave ---
// D# per CDNA5 ISA ch.8: group0 {count=1, lds_addr, global_addr, type=2},
// group1 {data_size=2B, tensor dims, tile dims, dim0 stride}.
__device__ __forceinline__ void tdm_load_2d(unsigned lds, const void* gaddr,
                                            unsigned tile_x, unsigned tile_y,
                                            unsigned tensor_x, unsigned tensor_y,
                                            unsigned stride_x) {
  unsigned long long ga = (unsigned long long)gaddr;
  u32x4 g0;
  g0[0] = 1u;                                           // count=1, no gather
  g0[1] = lds;                                          // lds_addr
  g0[2] = (unsigned)ga;                                 // global_addr[31:0]
  g0[3] = ((unsigned)(ga >> 32) & 0x1FFFFFFu) | (2u << 30);  // addr[56:32]|type=2
  u32x8 g1;
  g1[0] = 1u << 16;                                     // data_size=1 (2 bytes)
  g1[1] = (tensor_x & 0xFFFFu) << 16;                   // tensor_dim0[15:0]
  g1[2] = (tensor_x >> 16) | ((tensor_y & 0xFFFFu) << 16);
  g1[3] = (tensor_y >> 16) | (tile_x << 16);            // tile_dim0
  g1[4] = tile_y;                                       // tile_dim1 (tile_dim2=0)
  g1[5] = stride_x;                                     // tensor_dim0_stride lo
  g1[6] = 0u;
  g1[7] = 0u;
  asm volatile("tensor_load_to_lds %0, %1" ::"s"(g0), "s"(g1) : "memory");
}

// ---- WMMA fragment builders (ISA 7.12.2 layouts, wave32) -------------------
__device__ __forceinline__ bf16x16 load_frag_a(const unsigned short* src, int ld) {
  int lane = threadIdx.x & 31;
  int row = lane & 15;
  int c0 = (lane < 16) ? 0 : 8;
  bf16x16 f;
  unsigned short* p = reinterpret_cast<unsigned short*>(&f);
#pragma unroll
  for (int j = 0; j < 8; ++j) {
    p[j]     = src[row * ld + c0 + j];
    p[8 + j] = src[row * ld + c0 + 16 + j];
  }
  return f;
}
__device__ __forceinline__ bf16x16 load_frag_b(const unsigned short* srcT, int ld) {
  int lane = threadIdx.x & 31;
  int col = lane & 15;
  int k0 = (lane < 16) ? 0 : 8;
  bf16x16 f;
  unsigned short* p = reinterpret_cast<unsigned short*>(&f);
#pragma unroll
  for (int j = 0; j < 8; ++j) {
    p[j]     = srcT[col * ld + k0 + j];
    p[8 + j] = srcT[col * ld + k0 + 16 + j];
  }
  return f;
}

// ---- elementwise converts ---------------------------------------------------
__global__ __launch_bounds__(256) void cvt_bf16(const float* __restrict__ in,
                                                unsigned short* __restrict__ out,
                                                int n) {
  int i = blockIdx.x * 256 + threadIdx.x;
  if (i < n) out[i] = f2bf(in[i]);
}
__global__ __launch_bounds__(256) void cvt_bf16_T(const float* __restrict__ in,
                                                  unsigned short* __restrict__ out,
                                                  int R, int C) {
  int i = blockIdx.x * 256 + threadIdx.x;
  if (i < R * C) {
    int r = i / C, c = i - r * C;
    out[(size_t)c * R + r] = f2bf(in[i]);
  }
}

// ---- bf16 WMMA GEMM: C[M,N] f32 = A[M,K] * B (given as BT[N][K]) -----------
// Double-buffered LDS tiles staged by the Tensor Data Mover (wave 0 issues,
// TENSORcnt drained before the block barrier). BK=64 => 16 WMMA per wave/iter.
#define BM 128
#define BN 128
#define BKT 64
__global__ __launch_bounds__(256) void gemm_bf16_nt(
    const unsigned short* __restrict__ A, const unsigned short* __restrict__ BT,
    float* __restrict__ C, int M, int N, int Kd) {
  __shared__ __align__(16) unsigned short As[2][BM * BKT];
  __shared__ __align__(16) unsigned short Bs[2][BN * BKT];
  const int tid = threadIdx.x;
  const int wave = tid >> 5;
  const int wy = wave >> 1;  // 4 waves along M
  const int wx = wave & 1;   // 2 waves along N
  const int tm = blockIdx.y * BM;
  const int tn = blockIdx.x * BN;

  f32x8 acc[2][4];
#pragma unroll
  for (int i = 0; i < 2; ++i)
#pragma unroll
    for (int j = 0; j < 4; ++j) acc[i][j] = zero8();

  const int niter = Kd / BKT;
  if (wave == 0) {
    tdm_load_2d(lds_off(&As[0][0]), A + (size_t)tm * Kd, BKT, BM,
                (unsigned)Kd, (unsigned)M, (unsigned)Kd);
    tdm_load_2d(lds_off(&Bs[0][0]), BT + (size_t)tn * Kd, BKT, BN,
                (unsigned)Kd, (unsigned)N, (unsigned)Kd);
    __builtin_amdgcn_s_wait_tensorcnt(0);
  }
  __syncthreads();

  for (int it = 0; it < niter; ++it) {
    const int cur = it & 1, nxt = cur ^ 1;
    if (wave == 0 && it + 1 < niter) {  // prefetch next tiles while computing
      const int k0 = (it + 1) * BKT;
      tdm_load_2d(lds_off(&As[nxt][0]), A + (size_t)tm * Kd + k0, BKT, BM,
                  (unsigned)Kd, (unsigned)M, (unsigned)Kd);
      tdm_load_2d(lds_off(&Bs[nxt][0]), BT + (size_t)tn * Kd + k0, BKT, BN,
                  (unsigned)Kd, (unsigned)N, (unsigned)Kd);
    }
#pragma unroll
    for (int ks = 0; ks < BKT; ks += 32) {
      bf16x16 af0 = load_frag_a(&As[cur][(wy * 32) * BKT + ks], BKT);
      bf16x16 af1 = load_frag_a(&As[cur][(wy * 32 + 16) * BKT + ks], BKT);
#pragma unroll
      for (int nt = 0; nt < 4; ++nt) {
        bf16x16 bfv = load_frag_b(&Bs[cur][(wx * 64 + nt * 16) * BKT + ks], BKT);
        acc[0][nt] = wmma_bf16(af0, bfv, acc[0][nt]);
        acc[1][nt] = wmma_bf16(af1, bfv, acc[1][nt]);
      }
    }
    if (wave == 0 && it + 1 < niter) __builtin_amdgcn_s_wait_tensorcnt(0);
    __syncthreads();
  }

  const int lane = tid & 31;
  const int cn = lane & 15;
  const int mb = (lane < 16) ? 0 : 8;
#pragma unroll
  for (int mt = 0; mt < 2; ++mt)
#pragma unroll
    for (int nt = 0; nt < 4; ++nt) {
      float* cp = C + (size_t)(tm + wy * 32 + mt * 16) * N + tn + wx * 64 + nt * 16;
#pragma unroll
      for (int r = 0; r < 8; ++r) cp[(size_t)(mb + r) * N + cn] = acc[mt][nt][r];
    }
}

// ---- beta = sigmoid(X @ Wb), one block per token ---------------------------
__global__ __launch_bounds__(256) void beta_proj(const float* __restrict__ X,
                                                 const float* __restrict__ Wb,
                                                 float* __restrict__ beta) {
  const int t = blockIdx.x, tid = threadIdx.x;
  float acc[NH];
#pragma unroll
  for (int h = 0; h < NH; ++h) acc[h] = 0.f;
  for (int k = tid; k < HIDN; k += 256) {
    float xv = X[(size_t)t * HIDN + k];
#pragma unroll
    for (int h = 0; h < NH; ++h) acc[h] += xv * Wb[(size_t)k * NH + h];
  }
  __shared__ float red[NH][8];
  const int lane = tid & 31, wave = tid >> 5;
#pragma unroll
  for (int h = 0; h < NH; ++h) {
    float v = acc[h];
    for (int s = 16; s > 0; s >>= 1) v += __shfl_down(v, s);
    if (lane == 0) red[h][wave] = v;
  }
  __syncthreads();
  if (tid < NH) {
    float s = 0.f;
#pragma unroll
    for (int w = 0; w < 8; ++w) s += red[tid][w];
    beta[(size_t)t * NH + tid] = 1.f / (1.f + __expf(-s));
  }
}

// ---- depthwise causal conv(K=4) + SiLU (+ per-head l2norm) -> bf16 ---------
__global__ __launch_bounds__(256) void conv_silu(const float* __restrict__ P,
                                                 const float* __restrict__ cw,
                                                 unsigned short* __restrict__ out,
                                                 int mode) {
  const int t = blockIdx.x, tid = threadIdx.x;
  float y[8];
  float ss = 0.f;
#pragma unroll
  for (int e = 0; e < 8; ++e) {
    int c = tid * 8 + e;
    float a = 0.f;
#pragma unroll
    for (int j = 0; j < 4; ++j) {
      int tt = t + j - 3;
      if (tt >= 0) a += P[(size_t)tt * HIDN + c] * cw[c * 4 + j];
    }
    float s = a / (1.f + __expf(-a));  // silu
    y[e] = s;
    ss += s * s;
  }
  __shared__ float red[256];
  red[tid] = ss;
  __syncthreads();
  float r = 1.f;
  if (mode < 2) {
    int head = tid >> 4;
    float s = 0.f;
#pragma unroll
    for (int i = 0; i < 16; ++i) s += red[head * 16 + i];
    r = rsqrtf(s + 1e-6f);
    if (mode == 0) r *= 0.08838834764831845f;  // DK^-0.5
  }
#pragma unroll
  for (int e = 0; e < 8; ++e) out[(size_t)t * HIDN + tid * 8 + e] = f2bf(y[e] * r);
}

// ---- chunked delta-rule recurrence, one block (8 waves) per head -----------
#define CH 16
__global__ __launch_bounds__(256) void delta_rule_chunk(
    const unsigned short* __restrict__ qn, const unsigned short* __restrict__ kn,
    const unsigned short* __restrict__ vn, const float* __restrict__ beta,
    float* __restrict__ o) {
  const int h = blockIdx.x;
  const int tid = threadIdx.x;
  const int wave = tid >> 5;
  const int lane = tid & 31;

  __shared__ __align__(16) unsigned short qc[CH][DKK];
  __shared__ __align__(16) unsigned short kc[CH][DKK];
  __shared__ __align__(16) unsigned short kT[DKK][32];
  __shared__ __align__(16) unsigned short vT[DVV][32];
  __shared__ __align__(16) unsigned short SbT[DVV][DKK];
  __shared__ __align__(16) unsigned short wl[CH][DKK];
  __shared__ __align__(16) unsigned short Tmb[CH][32];
  __shared__ __align__(16) unsigned short qkb[CH][32];
  __shared__ float Af[CH][CH];
  __shared__ float betas[CH];

  for (int i = tid; i < DVV * DKK; i += 256) (&SbT[0][0])[i] = 0;
  for (int i = tid; i < DKK * 32; i += 256) { (&kT[0][0])[i] = 0; (&vT[0][0])[i] = 0; }
  for (int i = tid; i < CH * 32; i += 256) { (&Tmb[0][0])[i] = 0; (&qkb[0][0])[i] = 0; }

  f32x8 S[8];
#pragma unroll
  for (int m = 0; m < 8; ++m) S[m] = zero8();

  const int idx = tid * 8;
  const int lrow = idx >> 7;
  const int lcol = idx & 127;
  const unsigned lq = lds_off(&qc[lrow][lcol]);
  const unsigned lk = lds_off(&kc[lrow][lcol]);
  const int w16 = wave * 16;
  const int cn = lane & 15;
  const int mb = (lane < 16) ? 0 : 8;
  __syncthreads();

  for (int ci = 0; ci < T_TOK / CH; ++ci) {
    const int t0 = ci * CH;
    const size_t gbase = (size_t)(t0 + lrow) * HIDN + (size_t)h * DKK + lcol;
    async_b128(lq, qn + gbase);
    async_b128(lk, kn + gbase);
    unsigned short vr[8];
#pragma unroll
    for (int e = 0; e < 8; ++e) vr[e] = vn[gbase + e];
    if (tid < CH) betas[tid] = beta[(size_t)(t0 + tid) * NH + h];
    wait_async();
    __syncthreads();
#pragma unroll
    for (int e = 0; e < 8; ++e) {
      kT[lcol + e][lrow] = kc[lrow][lcol + e];
      vT[lcol + e][lrow] = vr[e];
    }
    __syncthreads();

    if (wave == 0) {
      f32x8 a = zero8();
#pragma unroll
      for (int s = 0; s < 4; ++s)
        a = wmma_bf16(load_frag_a(&kc[0][s * 32], DKK),
                      load_frag_b(&kc[0][s * 32], DKK), a);
#pragma unroll
      for (int r = 0; r < 8; ++r) {
        int m = mb + r;
        Af[m][cn] = (cn < m) ? (-betas[m] * a[r]) : 0.f;
      }
    } else if (wave == 1) {
      f32x8 a = zero8();
#pragma unroll
      for (int s = 0; s < 4; ++s)
        a = wmma_bf16(load_frag_a(&qc[0][s * 32], DKK),
                      load_frag_b(&kc[0][s * 32], DKK), a);
#pragma unroll
      for (int r = 0; r < 8; ++r) {
        int m = mb + r;
        qkb[m][cn] = f2bf((cn <= m) ? a[r] : 0.f);
      }
    }
    __syncthreads();

    if (wave == 0 && lane < 16) {
      for (int i = 1; i < CH; ++i) {
        float coef[CH];
#pragma unroll
        for (int j = 0; j < CH; ++j) coef[j] = Af[i][j];
        float v = coef[lane];
        for (int j = 0; j < i; ++j) v += coef[j] * Af[j][lane];
        Af[i][lane] = v;
      }
    }
    __syncthreads();
    {
      int i = tid >> 4, j = tid & 15;
      Tmb[i][j] = f2bf((Af[i][j] + (i == j ? 1.f : 0.f)) * betas[j]);
    }
    __syncthreads();

    bf16x16 tma = load_frag_a(&Tmb[0][0], 32);
    f32x8 wacc = wmma_bf16(tma, load_frag_b(&kT[w16][0], 32), zero8());
    f32x8 uacc = wmma_bf16(tma, load_frag_b(&vT[w16][0], 32), zero8());
#pragma unroll
    for (int r = 0; r < 8; ++r) wl[mb + r][w16 + cn] = f2bf(wacc[r]);
    __syncthreads();

    f32x8 ws_ = zero8();
#pragma unroll
    for (int s = 0; s < 4; ++s)
      ws_ = wmma_bf16(load_frag_a(&wl[0][s * 32], DKK),
                      load_frag_b(&SbT[w16][s * 32], DKK), ws_);
#pragma unroll
    for (int r = 0; r < 8; ++r) vT[w16 + cn][mb + r] = f2bf(uacc[r] - ws_[r]);
    __syncthreads();

    f32x8 oacc = zero8();
#pragma unroll
    for (int s = 0; s < 4; ++s)
      oacc = wmma_bf16(load_frag_a(&qc[0][s * 32], DKK),
                       load_frag_b(&SbT[w16][s * 32], DKK), oacc);
    oacc = wmma_bf16(load_frag_a(&qkb[0][0], 32), load_frag_b(&vT[w16][0], 32), oacc);
#pragma unroll
    for (int r = 0; r < 8; ++r)
      o[(size_t)(t0 + mb + r) * HIDN + (size_t)h * DVV + w16 + cn] = oacc[r];

    bf16x16 ub = load_frag_b(&vT[w16][0], 32);
#pragma unroll
    for (int mt = 0; mt < 8; ++mt) {
      S[mt] = wmma_bf16(load_frag_a(&kT[mt * 16][0], 32), ub, S[mt]);
#pragma unroll
      for (int r = 0; r < 8; ++r) SbT[w16 + cn][mt * 16 + mb + r] = f2bf(S[mt][r]);
    }
    __syncthreads();
  }
}

// ---- per-head RMSNorm on o, emit bf16 for the final GEMM -------------------
__global__ __launch_bounds__(256) void rmsnorm_o(const float* __restrict__ o,
                                                 const float* __restrict__ w,
                                                 unsigned short* __restrict__ out) {
  const int t = blockIdx.x, tid = threadIdx.x;
  float y[8];
  float ss = 0.f;
#pragma unroll
  for (int e = 0; e < 8; ++e) {
    int c = tid * 8 + e;
    float v = o[(size_t)t * HIDN + c];
    y[e] = v;
    ss += v * v;
  }
  __shared__ float red[256];
  red[tid] = ss;
  __syncthreads();
  int head = tid >> 4;
  float s = 0.f;
#pragma unroll
  for (int i = 0; i < 16; ++i) s += red[head * 16 + i];
  float r = rsqrtf(s * (1.f / 128.f) + 1e-6f);
#pragma unroll
  for (int e = 0; e < 8; ++e) {
    int c = tid * 8 + e;
    out[(size_t)t * HIDN + c] = f2bf(y[e] * r * w[c & 127]);
  }
}

extern "C" void kernel_launch(void* const* d_in, const int* in_sizes, int n_in,
                              void* d_out, int out_size, void* d_ws, size_t ws_size,
                              hipStream_t stream) {
  (void)in_sizes; (void)n_in; (void)out_size; (void)ws_size;
  const float* X   = (const float*)d_in[0];
  const float* Wq  = (const float*)d_in[1];
  const float* Wk  = (const float*)d_in[2];
  const float* Wv  = (const float*)d_in[3];
  const float* Wb  = (const float*)d_in[4];
  const float* cwq = (const float*)d_in[5];
  const float* cwk = (const float*)d_in[6];
  const float* cwv = (const float*)d_in[7];
  const float* onw = (const float*)d_in[8];
  const float* Wo  = (const float*)d_in[9];
  float* out = (float*)d_out;

  char* ws = (char*)d_ws;
  size_t off = 0;
  auto take = [&](size_t b) {
    char* p = ws + off;
    off += (b + 255) & ~(size_t)255;
    return p;
  };
  const size_t E = (size_t)T_TOK * HIDN;
  unsigned short* Xbf = (unsigned short*)take(E * 2);
  unsigned short* WqT = (unsigned short*)take(E * 2);
  unsigned short* WkT = (unsigned short*)take(E * 2);
  unsigned short* WvT = (unsigned short*)take(E * 2);
  unsigned short* WoT = (unsigned short*)take(E * 2);
  float* Pq = (float*)take(E * 4);
  float* Pk = (float*)take(E * 4);
  float* Pv = (float*)take(E * 4);
  float* betaB = (float*)take((size_t)T_TOK * NH * 4);
  unsigned short* qn = (unsigned short*)take(E * 2);
  unsigned short* kn = (unsigned short*)take(E * 2);
  unsigned short* vnb = (unsigned short*)take(E * 2);
  float* obuf = (float*)take(E * 4);
  unsigned short* Obf = (unsigned short*)take(E * 2);

  const int nblk = (int)(E / 256);
  cvt_bf16<<<nblk, 256, 0, stream>>>(X, Xbf, (int)E);
  cvt_bf16_T<<<nblk, 256, 0, stream>>>(Wq, WqT, HIDN, HIDN);
  cvt_bf16_T<<<nblk, 256, 0, stream>>>(Wk, WkT, HIDN, HIDN);
  cvt_bf16_T<<<nblk, 256, 0, stream>>>(Wv, WvT, HIDN, HIDN);
  cvt_bf16_T<<<nblk, 256, 0, stream>>>(Wo, WoT, HIDN, HIDN);

  dim3 gg(HIDN / BN, T_TOK / BM);
  gemm_bf16_nt<<<gg, 256, 0, stream>>>(Xbf, WqT, Pq, T_TOK, HIDN, HIDN);
  gemm_bf16_nt<<<gg, 256, 0, stream>>>(Xbf, WkT, Pk, T_TOK, HIDN, HIDN);
  gemm_bf16_nt<<<gg, 256, 0, stream>>>(Xbf, WvT, Pv, T_TOK, HIDN, HIDN);
  beta_proj<<<T_TOK, 256, 0, stream>>>(X, Wb, betaB);

  conv_silu<<<T_TOK, 256, 0, stream>>>(Pq, cwq, qn, 0);
  conv_silu<<<T_TOK, 256, 0, stream>>>(Pk, cwk, kn, 1);
  conv_silu<<<T_TOK, 256, 0, stream>>>(Pv, cwv, vnb, 2);

  delta_rule_chunk<<<NH, 256, 0, stream>>>(qn, kn, vnb, betaB, obuf);

  rmsnorm_o<<<T_TOK, 256, 0, stream>>>(obuf, onw, Obf);
  gemm_bf16_nt<<<gg, 256, 0, stream>>>(Obf, WoT, out, T_TOK, HIDN, HIDN);
}
